// RetNetForLEDGAR_29111288333019
// MI455X (gfx1250) — compile-verified, matrix operand order checked by reading, not run on hardware
//
#include <hip/hip_runtime.h>
#include <hip/hip_bf16.h>
#include <math.h>

typedef __attribute__((ext_vector_type(16))) _Float16 v16h;
typedef __attribute__((ext_vector_type(8)))  _Float16 v8h;
typedef __attribute__((ext_vector_type(8)))  float    v8f;
typedef __attribute__((ext_vector_type(4)))  int      v4i;

#define D_MODEL 768
#define N_LAYER 12
#define N_HEADS 12
#define DKH     64
#define DVH     128
#define D_VAL   1536
#define D_FFN   3072
#define BB      4
#define SS      1024
#define TT      (BB*SS)
#define NUM_LABELS 100
#define EPSF    1e-5f

// CDNA5 async global->LDS staging (ASYNCcnt path), guarded so the file
// compiles on toolchains without the builtin (falls back to sync staging).
#if defined(__AMDGCN__) && __has_builtin(__builtin_amdgcn_global_load_async_to_lds_b128)
#define USE_ASYNC_LDS 1
#else
#define USE_ASYNC_LDS 0
#endif

__device__ __forceinline__ void wait_async_all() {
#if USE_ASYNC_LDS
#if __has_builtin(__builtin_amdgcn_s_wait_asynccnt)
  __builtin_amdgcn_s_wait_asynccnt(0);
#else
  asm volatile("s_wait_asynccnt 0" ::: "memory");
#endif
#endif
}

// ------------------------- WMMA GEMM (TN form) -------------------------
// C[M,N](f32) = A[M,K](f16,row) * B[N,K](f16,row)^T ; batched via blockIdx.z
#define BM 128
#define BN 128
#define BKK 32
#define LDSS 40   // 32 + 8 halves pad; 80B row stride keeps 16B alignment

__device__ __forceinline__ v16h load_frag(const _Float16* base) {
  const int lane = threadIdx.x & 31;
  const int r = lane & 15, g = lane >> 4;
  const _Float16* p = base + r * LDSS + g * 8;
  v8h lo = *(const v8h*)(p);
  v8h hi = *(const v8h*)(p + 16);
  v16h f;
#pragma unroll
  for (int i = 0; i < 8; ++i) { f[i] = lo[i]; f[i + 8] = hi[i]; }
  return f;
}

__global__ __launch_bounds__(256)
void k_gemm_nt(const _Float16* __restrict__ A, const _Float16* __restrict__ B,
               float* __restrict__ C, int M, int N, int K,
               int lda, int ldb, int ldc,
               long long strA, long long strB, long long strC)
{
  __shared__ __align__(16) _Float16 As[2][BM * LDSS];
  __shared__ __align__(16) _Float16 Bs[2][BN * LDSS];

  A += (long long)blockIdx.z * strA;
  B += (long long)blockIdx.z * strB;
  C += (long long)blockIdx.z * strC;

  const int m0 = blockIdx.y * BM;
  const int n0 = blockIdx.x * BN;

  const int w  = threadIdx.x >> 5;  // 8 waves
  const int wm = w & 3;             // 4 wave-rows (32 C rows each)
  const int wn = w >> 2;            // 2 wave-cols (64 C cols each)

  v8f acc[2][4] = {};

  // stage one 128x32 tile of A and of B into LDS buffer `buf`
  auto stage = [&](int buf, int k0) {
#pragma unroll 2
    for (int i = threadIdx.x; i < (BM * BKK) / 8; i += 256) {
      const int r = (i * 8) / BKK, c = (i * 8) % BKK;
      const _Float16* ga = &A[(long long)(m0 + r) * lda + k0 + c];
      const _Float16* gb = &B[(long long)(n0 + r) * ldb + k0 + c];
      _Float16* la = &As[buf][r * LDSS + c];
      _Float16* lb = &Bs[buf][r * LDSS + c];
#if USE_ASYNC_LDS
      __builtin_amdgcn_global_load_async_to_lds_b128(
          (__attribute__((address_space(1))) v4i*)(void*)ga,
          (__attribute__((address_space(3))) v4i*)(void*)la, 0, 0);
      __builtin_amdgcn_global_load_async_to_lds_b128(
          (__attribute__((address_space(1))) v4i*)(void*)gb,
          (__attribute__((address_space(3))) v4i*)(void*)lb, 0, 0);
#else
      *(v8h*)la = *(const v8h*)ga;
      *(v8h*)lb = *(const v8h*)gb;
#endif
    }
  };

  stage(0, 0);
  int buf = 0;
  for (int k0 = 0; k0 < K; k0 += BKK) {
    wait_async_all();          // own-wave ASYNCcnt drained ...
    __syncthreads();           // ... then cross-wave visibility

    if (k0 + BKK < K) {
      stage(buf ^ 1, k0 + BKK);   // async DMA of next tile into other buffer
#if !USE_ASYNC_LDS
      __builtin_prefetch(&A[(long long)(m0 + (threadIdx.x >> 1)) * lda + k0 + BKK], 0, 1);
      __builtin_prefetch(&B[(long long)(n0 + (threadIdx.x >> 1)) * ldb + k0 + BKK], 0, 1);
#endif
    }

    v16h af[2], bf[4];
#pragma unroll
    for (int i = 0; i < 2; ++i) af[i] = load_frag(&As[buf][(wm * 32 + i * 16) * LDSS]);
#pragma unroll
    for (int j = 0; j < 4; ++j) bf[j] = load_frag(&Bs[buf][(wn * 64 + j * 16) * LDSS]);

#pragma unroll
    for (int i = 0; i < 2; ++i)
#pragma unroll
      for (int j = 0; j < 4; ++j)
        acc[i][j] = __builtin_amdgcn_wmma_f32_16x16x32_f16(
            false, af[i], false, bf[j], (short)0, acc[i][j], false, false);

    buf ^= 1;
  }

  const int lane = threadIdx.x & 31;
  const int cn   = lane & 15;
  const int half = lane >> 4;
#pragma unroll
  for (int i = 0; i < 2; ++i)
#pragma unroll
    for (int j = 0; j < 4; ++j) {
      float* cp = C + (long long)(m0 + wm * 32 + i * 16 + half * 8) * ldc
                    + (n0 + wn * 64 + j * 16 + cn);
#pragma unroll
      for (int r = 0; r < 8; ++r) cp[(long long)r * ldc] = acc[i][j][r];
    }
}

// ------------------------- elementwise / reduction kernels -------------------------

template <typename OutT>
__global__ __launch_bounds__(256)
void k_layernorm(const float* __restrict__ x, const float* __restrict__ g,
                 const float* __restrict__ b, OutT* __restrict__ out)
{
  __shared__ float red[256];
  const long long row = blockIdx.x;
  const float* xr = x + row * D_MODEL;

  float s = 0.f;
  for (int d = threadIdx.x; d < D_MODEL; d += 256) s += xr[d];
  red[threadIdx.x] = s; __syncthreads();
  for (int st = 128; st > 0; st >>= 1) {
    if ((int)threadIdx.x < st) red[threadIdx.x] += red[threadIdx.x + st];
    __syncthreads();
  }
  const float mu = red[0] / (float)D_MODEL;
  __syncthreads();

  float v = 0.f;
  for (int d = threadIdx.x; d < D_MODEL; d += 256) { float t = xr[d] - mu; v += t * t; }
  red[threadIdx.x] = v; __syncthreads();
  for (int st = 128; st > 0; st >>= 1) {
    if ((int)threadIdx.x < st) red[threadIdx.x] += red[threadIdx.x + st];
    __syncthreads();
  }
  const float rs = rsqrtf(red[0] / (float)D_MODEL + EPSF);

  for (int d = threadIdx.x; d < D_MODEL; d += 256)
    out[row * D_MODEL + d] = (OutT)((xr[d] - mu) * rs * g[d] + b[d]);
}

__global__ void k_sincos(float* __restrict__ sinb, float* __restrict__ cosb) {
  const int idx = blockIdx.x * blockDim.x + threadIdx.x;
  if (idx >= SS * DKH) return;
  const int s = idx / DKH, d = idx % DKH;
  const float ex   = (float)(d >> 1) / 31.0f;        // linspace(0,1,32)
  const float freq = powf(10000.f, -ex);
  const float a    = (float)s * freq;
  sinb[idx] = sinf(a);
  cosb[idx] = cosf(a);
}

__global__ void k_rnorm(float* __restrict__ rn) {
  const int idx = blockIdx.x * blockDim.x + threadIdx.x;
  if (idx >= N_HEADS * SS) return;
  const int h = idx / SS, n = idx % SS;
  const float gamma = 1.f - exp2f(-5.f - (float)h);
  const float sum   = (1.f - powf(gamma, (float)(n + 1))) / (1.f - gamma);
  rn[idx] = rsqrtf(sum);
}

// theta_shift + scale, scatter into [b][h][s][dk] f16
__global__ void k_rope(const float* __restrict__ q, const float* __restrict__ sinb,
                       const float* __restrict__ cosb, _Float16* __restrict__ out,
                       float scale)
{
  const long long idx = (long long)blockIdx.x * blockDim.x + threadIdx.x;
  if (idx >= (long long)TT * D_MODEL) return;
  const int c = (int)(idx % D_MODEL);
  const int d = c % DKH;
  const int h = c / DKH;
  const long long t = idx / D_MODEL;
  const int s = (int)(t % SS);
  const int b = (int)(t / SS);
  const float v = q[idx];
  const float r = (d & 1) ? q[idx - 1] : -q[idx + 1];
  const float o = (v * cosb[s * DKH + d] + r * sinb[s * DKH + d]) * scale;
  out[(((long long)(b * N_HEADS + h)) * SS + s) * DKH + d] = (_Float16)o;
}

// v (T,D_VAL) f32 -> vt [b][h][dv][s] f16 (transposed for TN gemm)
__global__ void k_vtrans(const float* __restrict__ vf, _Float16* __restrict__ vt) {
  const long long idx = (long long)blockIdx.x * blockDim.x + threadIdx.x;
  if (idx >= (long long)BB * N_HEADS * DVH * SS) return;
  const int s  = (int)(idx % SS);
  const int dv = (int)((idx / SS) % DVH);
  const long long z = idx / ((long long)SS * DVH);
  const int h = (int)(z % N_HEADS), b = (int)(z / N_HEADS);
  vt[idx] = (_Float16)vf[((long long)(b * SS + s)) * D_VAL + h * DVH + dv];
}

// fused: apply decay mask D(n,m) on the fly, row-abs-sum normalize, emit f16 P
__global__ __launch_bounds__(256)
void k_decay_norm(const float* __restrict__ qk, _Float16* __restrict__ p,
                  const float* __restrict__ rn)
{
  __shared__ float red[256];
  const long long row = blockIdx.x;                 // 0 .. B*H*S-1
  const int n = (int)(row % SS);
  const int h = (int)((row / SS) % N_HEADS);
  const float lg  = logf(1.f - exp2f(-5.f - (float)h));
  const float rnv = rn[h * SS + n];
  const float* qr = qk + row * SS;

  float vals[4];
  float asum = 0.f;
#pragma unroll
  for (int j = 0; j < 4; ++j) {
    const int m = (int)threadIdx.x + j * 256;
    float v = qr[m];
    v = (n >= m) ? v * expf((float)(n - m) * lg) * rnv : 0.f;
    vals[j] = v;
    asum += fabsf(v);
  }
  red[threadIdx.x] = asum; __syncthreads();
  for (int st = 128; st > 0; st >>= 1) {
    if ((int)threadIdx.x < st) red[threadIdx.x] += red[threadIdx.x + st];
    __syncthreads();
  }
  const float scale = 1.f / fmaxf(red[0], 1.f);
#pragma unroll
  for (int j = 0; j < 4; ++j)
    p[row * SS + threadIdx.x + j * 256] = (_Float16)(vals[j] * scale);
}

// rms-normalize o over DV then gate by silu(g); scatter into (B,S,D_VAL) f16
__global__ void k_rmsgate(const float* __restrict__ o, const float* __restrict__ g,
                          _Float16* __restrict__ out)
{
  const long long row = (long long)blockIdx.x * blockDim.x + threadIdx.x;
  if (row >= (long long)BB * N_HEADS * SS) return;
  const int s = (int)(row % SS);
  const long long z = row / SS;
  const int h = (int)(z % N_HEADS), b = (int)(z / N_HEADS);
  const float* op = o + row * DVH;
  float ss = 0.f;
  for (int d = 0; d < DVH; ++d) ss += op[d] * op[d];
  const float r = rsqrtf(ss / (float)DVH + EPSF);
  const long long base = ((long long)(b * SS + s)) * D_VAL + h * DVH;
  for (int d = 0; d < DVH; ++d) {
    const float gv  = g[base + d];
    const float sig = gv / (1.f + expf(-gv));
    out[base + d] = (_Float16)(sig * op[d] * r);
  }
}

__global__ void k_add(float* __restrict__ x, const float* __restrict__ y, long long n) {
  const long long i = (long long)blockIdx.x * blockDim.x + threadIdx.x;
  if (i < n) x[i] += y[i];
}

__global__ void k_silumul(const float* __restrict__ a, const float* __restrict__ b,
                          _Float16* __restrict__ out, long long n) {
  const long long i = (long long)blockIdx.x * blockDim.x + threadIdx.x;
  if (i >= n) return;
  const float av = a[i];
  out[i] = (_Float16)((av / (1.f + expf(-av))) * b[i]);
}

// fp32 W(K,N) -> f16 W^T(N,K)
__global__ void k_cvt_t(const float* __restrict__ w, _Float16* __restrict__ wt,
                        int K, int N) {
  const long long idx = (long long)blockIdx.x * blockDim.x + threadIdx.x;
  if (idx >= (long long)K * N) return;
  const int n = (int)(idx % N);
  const long long k = idx / N;
  wt[(long long)n * K + k] = (_Float16)w[idx];
}

__global__ void k_embed(const int* __restrict__ ids, const float* __restrict__ emb,
                        float* __restrict__ x) {
  const long long idx = (long long)blockIdx.x * blockDim.x + threadIdx.x;
  if (idx >= (long long)TT * D_MODEL) return;
  const long long t = idx / D_MODEL;
  const int d = (int)(idx % D_MODEL);
  x[idx] = emb[(long long)ids[t] * D_MODEL + d];
}

__global__ void k_meanpool(const float* __restrict__ xf, float* __restrict__ pooled) {
  const int idx = blockIdx.x * blockDim.x + threadIdx.x;
  if (idx >= BB * D_MODEL) return;
  const int b = idx / D_MODEL, d = idx % D_MODEL;
  float s = 0.f;
  for (int t = 0; t < SS; ++t) s += xf[((long long)(b * SS + t)) * D_MODEL + d];
  pooled[idx] = s / (float)SS;
}

__global__ void k_classifier(const float* __restrict__ pooled, const float* __restrict__ w,
                             const float* __restrict__ bias, float* __restrict__ out) {
  const int idx = blockIdx.x * blockDim.x + threadIdx.x;
  if (idx >= BB * NUM_LABELS) return;
  const int b = idx / NUM_LABELS, l = idx % NUM_LABELS;
  float s = bias[l];
  for (int d = 0; d < D_MODEL; ++d)
    s += pooled[b * D_MODEL + d] * w[d * NUM_LABELS + l];
  out[idx] = s;
}

// ------------------------- host orchestration -------------------------

extern "C" void kernel_launch(void* const* d_in, const int* in_sizes, int n_in,
                              void* d_out, int out_size, void* d_ws, size_t ws_size,
                              hipStream_t stream)
{
  (void)in_sizes; (void)n_in; (void)out_size; (void)ws_size;

  const int*   ids  = (const int*)  d_in[0];
  const float* emb  = (const float*)d_in[1];
  const float* wq   = (const float*)d_in[2];
  const float* wk   = (const float*)d_in[3];
  const float* wv   = (const float*)d_in[4];
  const float* wg   = (const float*)d_in[5];
  const float* wo   = (const float*)d_in[6];
  const float* ln1g = (const float*)d_in[7];
  const float* ln1b = (const float*)d_in[8];
  const float* ln2g = (const float*)d_in[9];
  const float* ln2b = (const float*)d_in[10];
  const float* fc1  = (const float*)d_in[11];
  const float* gate = (const float*)d_in[12];
  const float* fc2  = (const float*)d_in[13];
  const float* lnfg = (const float*)d_in[14];
  const float* lnfb = (const float*)d_in[15];
  const float* clfw = (const float*)d_in[16];
  const float* clfb = (const float*)d_in[17];
  float* out = (float*)d_out;

  char* base = (char*)d_ws;
  size_t off = 0;
  auto alloc = [&](size_t bytes) -> void* {
    void* p = base + off;
    off += (bytes + 255) & ~(size_t)255;
    return p;
  };
  const size_t H2 = sizeof(_Float16);

  _Float16* wqT = (_Float16*)alloc((size_t)N_LAYER * D_MODEL * D_MODEL * H2);
  _Float16* wkT = (_Float16*)alloc((size_t)N_LAYER * D_MODEL * D_MODEL * H2);
  _Float16* wvT = (_Float16*)alloc((size_t)N_LAYER * D_MODEL * D_VAL  * H2);
  _Float16* wgT = (_Float16*)alloc((size_t)N_LAYER * D_MODEL * D_VAL  * H2);
  _Float16* woT = (_Float16*)alloc((size_t)N_LAYER * D_VAL  * D_MODEL * H2);
  _Float16* f1T = (_Float16*)alloc((size_t)N_LAYER * D_MODEL * D_FFN  * H2);
  _Float16* gtT = (_Float16*)alloc((size_t)N_LAYER * D_MODEL * D_FFN  * H2);
  _Float16* f2T = (_Float16*)alloc((size_t)N_LAYER * D_FFN  * D_MODEL * H2);

  float* sinb = (float*)alloc((size_t)SS * DKH * 4);
  float* cosb = (float*)alloc((size_t)SS * DKH * 4);
  float* rn   = (float*)alloc((size_t)N_HEADS * SS * 4);

  float*    x    = (float*)   alloc((size_t)TT * D_MODEL * 4);
  _Float16* h16  = (_Float16*)alloc((size_t)TT * D_MODEL * H2);
  float*    qf   = (float*)   alloc((size_t)TT * D_MODEL * 4);
  float*    kf   = (float*)   alloc((size_t)TT * D_MODEL * 4);
  float*    vf   = (float*)   alloc((size_t)TT * D_VAL  * 4);
  float*    gf   = (float*)   alloc((size_t)TT * D_VAL  * 4);
  _Float16* qr   = (_Float16*)alloc((size_t)BB * N_HEADS * SS * DKH * H2);
  _Float16* kr   = (_Float16*)alloc((size_t)BB * N_HEADS * SS * DKH * H2);
  _Float16* vt   = (_Float16*)alloc((size_t)BB * N_HEADS * DVH * SS * H2);
  float*    qk   = (float*)   alloc((size_t)BB * N_HEADS * SS * SS * 4);
  _Float16* p16  = (_Float16*)alloc((size_t)BB * N_HEADS * SS * SS * H2);
  float*    obuf = (float*)   alloc((size_t)BB * N_HEADS * SS * DVH * 4);
  _Float16* og   = (_Float16*)alloc((size_t)TT * D_VAL * H2);
  float*    tmpO = (float*)   alloc((size_t)TT * D_MODEL * 4);
  float*    ffa  = (float*)   alloc((size_t)TT * D_FFN * 4);
  float*    ffb  = (float*)   alloc((size_t)TT * D_FFN * 4);
  _Float16* prod = (_Float16*)alloc((size_t)TT * D_FFN * H2);
  float*    xf   = (float*)   alloc((size_t)TT * D_MODEL * 4);
  float*    pooled = (float*) alloc((size_t)BB * D_MODEL * 4);

  auto cvt = [&](const float* w, _Float16* wt, int K, int N) {
    const long long n = (long long)K * N;
    k_cvt_t<<<(unsigned)((n + 255) / 256), 256, 0, stream>>>(w, wt, K, N);
  };
  auto gemm = [&](const _Float16* A, const _Float16* B, float* C,
                  int M, int N, int K, int lda, int ldb, int ldc,
                  long long sA, long long sB, long long sC, int batch) {
    dim3 g((unsigned)(N / BN), (unsigned)(M / BM), (unsigned)batch);
    k_gemm_nt<<<g, 256, 0, stream>>>(A, B, C, M, N, K, lda, ldb, ldc, sA, sB, sC);
  };

  // weight conversion (fp32 -> f16 transposed), all layers
  for (int l = 0; l < N_LAYER; ++l) {
    cvt(wq  + (size_t)l * D_MODEL * D_MODEL, wqT + (size_t)l * D_MODEL * D_MODEL, D_MODEL, D_MODEL);
    cvt(wk  + (size_t)l * D_MODEL * D_MODEL, wkT + (size_t)l * D_MODEL * D_MODEL, D_MODEL, D_MODEL);
    cvt(wv  + (size_t)l * D_MODEL * D_VAL,   wvT + (size_t)l * D_MODEL * D_VAL,   D_MODEL, D_VAL);
    cvt(wg  + (size_t)l * D_MODEL * D_VAL,   wgT + (size_t)l * D_MODEL * D_VAL,   D_MODEL, D_VAL);
    cvt(wo  + (size_t)l * D_VAL * D_MODEL,   woT + (size_t)l * D_VAL * D_MODEL,   D_VAL,   D_MODEL);
    cvt(fc1 + (size_t)l * D_MODEL * D_FFN,   f1T + (size_t)l * D_MODEL * D_FFN,   D_MODEL, D_FFN);
    cvt(gate+ (size_t)l * D_MODEL * D_FFN,   gtT + (size_t)l * D_MODEL * D_FFN,   D_MODEL, D_FFN);
    cvt(fc2 + (size_t)l * D_FFN * D_MODEL,   f2T + (size_t)l * D_FFN * D_MODEL,   D_FFN,   D_MODEL);
  }

  k_sincos<<<(SS * DKH + 255) / 256, 256, 0, stream>>>(sinb, cosb);
  k_rnorm<<<(N_HEADS * SS + 255) / 256, 256, 0, stream>>>(rn);
  {
    const long long n = (long long)TT * D_MODEL;
    k_embed<<<(unsigned)((n + 255) / 256), 256, 0, stream>>>(ids, emb, x);
  }

  const long long nTD  = (long long)TT * D_MODEL;
  const long long nVT  = (long long)BB * N_HEADS * DVH * SS;
  const long long nRow = (long long)BB * N_HEADS * SS;
  const long long nF   = (long long)TT * D_FFN;

  for (int l = 0; l < N_LAYER; ++l) {
    const _Float16* wqTl = wqT + (size_t)l * D_MODEL * D_MODEL;
    const _Float16* wkTl = wkT + (size_t)l * D_MODEL * D_MODEL;
    const _Float16* wvTl = wvT + (size_t)l * D_MODEL * D_VAL;
    const _Float16* wgTl = wgT + (size_t)l * D_MODEL * D_VAL;
    const _Float16* woTl = woT + (size_t)l * D_VAL * D_MODEL;
    const _Float16* f1Tl = f1T + (size_t)l * D_MODEL * D_FFN;
    const _Float16* gtTl = gtT + (size_t)l * D_MODEL * D_FFN;
    const _Float16* f2Tl = f2T + (size_t)l * D_FFN * D_MODEL;

    // ---- retention block ----
    k_layernorm<_Float16><<<TT, 256, 0, stream>>>(x, ln1g + l * D_MODEL, ln1b + l * D_MODEL, h16);
    gemm(h16, wqTl, qf, TT, D_MODEL, D_MODEL, D_MODEL, D_MODEL, D_MODEL, 0, 0, 0, 1);
    gemm(h16, wkTl, kf, TT, D_MODEL, D_MODEL, D_MODEL, D_MODEL, D_MODEL, 0, 0, 0, 1);
    gemm(h16, wvTl, vf, TT, D_VAL,  D_MODEL, D_MODEL, D_MODEL, D_VAL,  0, 0, 0, 1);
    gemm(h16, wgTl, gf, TT, D_VAL,  D_MODEL, D_MODEL, D_MODEL, D_VAL,  0, 0, 0, 1);

    k_rope<<<(unsigned)((nTD + 255) / 256), 256, 0, stream>>>(qf, sinb, cosb, qr, 1.0f);
    k_rope<<<(unsigned)((nTD + 255) / 256), 256, 0, stream>>>(kf, sinb, cosb, kr, 0.125f); // DK^-0.5
    k_vtrans<<<(unsigned)((nVT + 255) / 256), 256, 0, stream>>>(vf, vt);

    // qk = qr @ kr^T  per (b,h):  M=N=1024, K=64
    gemm(qr, kr, qk, SS, SS, DKH, DKH, DKH, SS,
         (long long)SS * DKH, (long long)SS * DKH, (long long)SS * SS, BB * N_HEADS);
    k_decay_norm<<<(unsigned)nRow, 256, 0, stream>>>(qk, p16, rn);
    // o = P @ V  per (b,h): M=1024, N=128, K=1024 (vt is V^T in (DV,S))
    gemm(p16, vt, obuf, SS, DVH, SS, SS, SS, DVH,
         (long long)SS * SS, (long long)DVH * SS, (long long)SS * DVH, BB * N_HEADS);

    k_rmsgate<<<(unsigned)((nRow + 255) / 256), 256, 0, stream>>>(obuf, gf, og);
    gemm(og, woTl, tmpO, TT, D_MODEL, D_VAL, D_VAL, D_VAL, D_MODEL, 0, 0, 0, 1);
    k_add<<<(unsigned)((nTD + 255) / 256), 256, 0, stream>>>(x, tmpO, nTD);

    // ---- gated FFN ----
    k_layernorm<_Float16><<<TT, 256, 0, stream>>>(x, ln2g + l * D_MODEL, ln2b + l * D_MODEL, h16);
    gemm(h16, f1Tl, ffa, TT, D_FFN, D_MODEL, D_MODEL, D_MODEL, D_FFN, 0, 0, 0, 1);
    gemm(h16, gtTl, ffb, TT, D_FFN, D_MODEL, D_MODEL, D_MODEL, D_FFN, 0, 0, 0, 1);
    k_silumul<<<(unsigned)((nF + 255) / 256), 256, 0, stream>>>(ffa, ffb, prod, nF);
    gemm(prod, f2Tl, tmpO, TT, D_MODEL, D_FFN, D_FFN, D_FFN, D_MODEL, 0, 0, 0, 1);
    k_add<<<(unsigned)((nTD + 255) / 256), 256, 0, stream>>>(x, tmpO, nTD);
  }

  // ---- head ----
  k_layernorm<float><<<TT, 256, 0, stream>>>(x, lnfg, lnfb, xf);
  k_meanpool<<<(BB * D_MODEL + 255) / 256, 256, 0, stream>>>(xf, pooled);
  k_classifier<<<(BB * NUM_LABELS + 255) / 256, 256, 0, stream>>>(pooled, clfw, clfb, out);
}
